// Batched_STHD_SpGAT_75814762709167
// MI455X (gfx1250) — matrix operand order repaired and checked by reading, not compiled, not run on hardware
//
#include <hip/hip_runtime.h>
#include <math.h>

typedef __attribute__((ext_vector_type(16))) _Float16 v16h;
typedef __attribute__((ext_vector_type(8)))  float    v8f;

#define N_SUB       50000
#define NUM_CELLS   100000
#define NUM_CLASSES 32
#define NUM_GENES   256
#define NUM_EDGES   400000
#define GAT_OUT     8
#define NEG_SLOPE   0.2f
#define NTILES      (N_SUB / 16)   /* 3125 exactly */
#define TAB_HALVES  20480          /* invvB(8192) + muvB(8192) + lrB(4096) = 40KB */

// ---------- helpers ----------
__device__ inline float waveSum(float v) {
  #pragma unroll
  for (int s = 16; s > 0; s >>= 1) v += __shfl_xor(v, s, 32);
  return v;
}
__device__ inline float waveMax(float v) {
  #pragma unroll
  for (int s = 16; s > 0; s >>= 1) v = fmaxf(v, __shfl_xor(v, s, 32));
  return v;
}
// order-preserving float <-> uint key for atomicMax on floats
__device__ inline unsigned fkey(float f) {
  unsigned u = __float_as_uint(f);
  return (u & 0x80000000u) ? ~u : (u | 0x80000000u);
}
__device__ inline float keyf(unsigned k) {
  unsigned u = (k & 0x80000000u) ? (k & 0x7FFFFFFFu) : ~k;
  return __uint_as_float(u);
}

// ---------- 0: zero scratch ----------
__global__ void k_zero(float* p, int n) {
  int i = blockIdx.x * blockDim.x + threadIdx.x;
  if (i < n) p[i] = 0.0f;
}

// ---------- 1: pre-pack WMMA B-operand tables (f16), D[c], b-vector ----------
// B k order chosen here (contiguous, matches A) for V_WMMA_F32_16X16X32_F16: lane<16 -> col N=lane, K=0..15
// in elements 0..15; lane>=16 -> col N=lane-16, K=16..31.
// Table index: ((kb*NT + nt)*32 + lane)*16 + e  with k = kb*32 + (lane>>4)*16 + e.
__global__ void k_build_tabs(const float* Mu, const float* Var,
                             const float* W_l, const float* W_r,
                             const float* b_l, const float* b_r,
                             _Float16* gtab, float* Dvec, float* bvec) {
  _Float16* invvB = gtab;
  _Float16* muvB  = gtab + 8192;
  _Float16* lrB   = gtab + 16384;
  int i = blockIdx.x * blockDim.x + threadIdx.x;
  if (i < 8192) {                       // 8 kb * 2 nt * 32 lanes * 16 elems
    int e = i & 15, lane = (i >> 4) & 31, nt = (i >> 9) & 1, kb = i >> 10;
    int k = kb * 32 + ((lane >> 4) << 4) + e;
    int c = (nt << 4) + (lane & 15);
    float iv = 1.0f / Var[c * NUM_GENES + k];
    invvB[i] = (_Float16)iv;
    muvB[i]  = (_Float16)(Mu[c * NUM_GENES + k] * iv);
  }
  if (i < 4096) {                       // 8 kb * 1 nt * 32 * 16 : cols 0-7=W_l, 8-15=W_r
    int e = i & 15, lane = (i >> 4) & 31, kb = i >> 9;
    int k = kb * 32 + ((lane >> 4) << 4) + e;
    int n = lane & 15;
    float v = (n < 8) ? W_l[k * GAT_OUT + n] : W_r[k * GAT_OUT + (n - 8)];
    lrB[i] = (_Float16)v;
  }
  if (i < NUM_CLASSES) {                // D[c] = sum_g Mu^2/Var
    float s = 0.0f;
    for (int g = 0; g < NUM_GENES; ++g) {
      float m = Mu[i * NUM_GENES + g];
      s += m * m / Var[i * NUM_GENES + g];
    }
    Dvec[i] = s;
  }
  if (i >= 32 && i < 48) {
    int j = i - 32;
    bvec[j] = (j < 8) ? b_l[j] : b_r[j - 8];
  }
}

// ---------- 2: spillover accumulation (one wave per edge) ----------
__global__ void k_spill_acc(const float* __restrict__ x, const int* __restrict__ ei,
                            float* acc, float* deg) {
  int gid = blockIdx.x * blockDim.x + threadIdx.x;
  int e = gid >> 5, lane = gid & 31;
  if (e >= NUM_EDGES) return;
  int row = ei[e], col = ei[NUM_EDGES + e];
  if (lane == 0) atomicAdd(&deg[row], 1.0f);
  const float* xs = x + (size_t)col * NUM_GENES;
  float* ad = acc + (size_t)row * NUM_GENES;
  #pragma unroll
  for (int g = lane; g < NUM_GENES; g += 32) atomicAdd(&ad[g], xs[g]);
}

// ---------- 3: x_clean = relu(x - rate_c * acc/deg), in place over acc ----------
__global__ void k_xclean(const float* __restrict__ x, const float* __restrict__ deg,
                         const float* rate, float* acc) {
  int i = blockIdx.x * blockDim.x + threadIdx.x;
  if (i >= N_SUB * NUM_GENES) return;
  int r = i >> 8;                       // / NUM_GENES
  float rc = fminf(fmaxf(rate[0], 0.0f), 0.2f);
  float d = fmaxf(deg[r], 1.0f);
  acc[i] = fmaxf(x[i] - rc * (acc[i] / d), 0.0f);
}

// ---------- 4: P_sub = softmax(W[subset], axis=1) (one wave per row) ----------
__global__ void k_softmax(const float* __restrict__ W, const int* __restrict__ subset,
                          float* __restrict__ P) {
  int gid = blockIdx.x * blockDim.x + threadIdx.x;
  int row = gid >> 5, lane = gid & 31;
  if (row >= N_SUB) return;
  float w = W[(size_t)subset[row] * NUM_CLASSES + lane];
  float m = waveMax(w);
  float ex = expf(w - m);
  float den = waveSum(ex);
  P[(size_t)row * NUM_CLASSES + lane] = ex / den;
}

// ---------- 5: WMMA GEMMs + fused F_c / ll_prot reduction ----------
// B-operand tables are staged once per block into LDS with the gfx1250 async
// Global->LDS data mover (GLOBAL_LOAD_ASYNC_TO_LDS_B128, ASYNCcnt), freeing the
// vector-load pipe for the streaming x_clean reads inside the K-loop.
__global__ __launch_bounds__(128) void k_wmma(const float* __restrict__ xc,
    const _Float16* __restrict__ gtab, const float* __restrict__ Dvec,
    const float* __restrict__ bvec, const int* __restrict__ subset,
    const float* __restrict__ S, const float* __restrict__ P,
    float* __restrict__ xlr, float* llacc) {
  __shared__ __align__(32) _Float16 ldsTab[TAB_HALVES];

  // Cooperative async fill: 40960 B = 2560 x 16B chunks over 128 threads.
  {
    unsigned ldsbase = (unsigned)(uintptr_t)(&ldsTab[0]);   // addr[31:0] = LDS offset
    const char* g = (const char*)gtab;
    for (int i = threadIdx.x; i < TAB_HALVES / 8; i += 128) {
      unsigned loff = ldsbase + (unsigned)i * 16u;
      const void* ga = (const void*)(g + (size_t)i * 16u);
      asm volatile("global_load_async_to_lds_b128 %0, %1, off"
                   :: "v"(loff), "v"(ga) : "memory");
    }
    asm volatile("s_wait_asynccnt 0x0" ::: "memory");
  }
  __syncthreads();

  int wave = threadIdx.x >> 5;
  int lane = threadIdx.x & 31;
  int tile = blockIdx.x * 4 + wave;
  if (tile < NTILES) {                  // wave-uniform: EXEC stays all-ones for WMMA
    int half = lane >> 4;
    const float* xrow = xc + (size_t)(tile * 16 + (lane & 15)) * NUM_GENES;
    const _Float16* linvv = &ldsTab[0];
    const _Float16* lmuv  = &ldsTab[8192];
    const _Float16* llr   = &ldsTab[16384];
    v8f accA0 = {}, accA1 = {}, accB0 = {}, accB1 = {}, accX = {};
    #pragma unroll 2
    for (int kb = 0; kb < 8; ++kb) {
      // A-operand k order used here (contiguous, matches B tables): lane<16 e0..15->K0..15;
      // lane>=16 e0..15->K16..31 (within this kb's 32-chunk).
      const float* xp = xrow + kb * 32 + half * 16;
      float4 f0 = *(const float4*)(xp);
      float4 f1 = *(const float4*)(xp + 4);
      float4 f2 = *(const float4*)(xp + 8);
      float4 f3 = *(const float4*)(xp + 12);
      float vals[16] = { f0.x,f0.y,f0.z,f0.w, f1.x,f1.y,f1.z,f1.w,
                         f2.x,f2.y,f2.z,f2.w, f3.x,f3.y,f3.z,f3.w };
      v16h a, a2;
      #pragma unroll
      for (int t = 0; t < 16; ++t) {
        a[t]  = (_Float16)vals[t];
        a2[t] = (_Float16)(vals[t] * vals[t]);
      }
      v16h bA0 = *(const v16h*)(linvv + (size_t)((kb * 2 + 0) * 32 + lane) * 16);
      v16h bA1 = *(const v16h*)(linvv + (size_t)((kb * 2 + 1) * 32 + lane) * 16);
      v16h bB0 = *(const v16h*)(lmuv  + (size_t)((kb * 2 + 0) * 32 + lane) * 16);
      v16h bB1 = *(const v16h*)(lmuv  + (size_t)((kb * 2 + 1) * 32 + lane) * 16);
      v16h bX  = *(const v16h*)(llr   + (size_t)(kb * 32 + lane) * 16);
      accA0 = __builtin_amdgcn_wmma_f32_16x16x32_f16(false, a2, false, bA0, (short)0, accA0, false, false);
      accA1 = __builtin_amdgcn_wmma_f32_16x16x32_f16(false, a2, false, bA1, (short)0, accA1, false, false);
      accB0 = __builtin_amdgcn_wmma_f32_16x16x32_f16(false, a,  false, bB0, (short)0, accB0, false, false);
      accB1 = __builtin_amdgcn_wmma_f32_16x16x32_f16(false, a,  false, bB1, (short)0, accB1, false, false);
      accX  = __builtin_amdgcn_wmma_f32_16x16x32_f16(false, a,  false, bX,  (short)0, accX,  false, false);
    }
    // C/D layout: col N = lane&15; row M = v + 8*(lane>=16)
    int col = lane & 15;
    float part = 0.0f;
    #pragma unroll
    for (int v = 0; v < 8; ++v) {
      int m = tile * 16 + v + 8 * half;
      xlr[(size_t)m * 16 + col] = accX[v] + bvec[col];
      float Sv = S[subset[m]];
      float F0 = -0.5f * (accA0[v] - 2.0f * Sv * accB0[v] + Sv * Sv * Dvec[col]);
      float F1 = -0.5f * (accA1[v] - 2.0f * Sv * accB1[v] + Sv * Sv * Dvec[16 + col]);
      part += P[(size_t)m * 32 + col] * F0 + P[(size_t)m * 32 + 16 + col] * F1;
    }
    part = waveSum(part);
    if (lane == 0) atomicAdd(llacc, part);
  }
}

// ---------- 6: per-edge attention logits + segment max over dst ----------
__global__ void k_edge(const int* __restrict__ ei, const float* __restrict__ xlr,
                       const float* __restrict__ att, float* ev, unsigned* mkey) {
  int e = blockIdx.x * blockDim.x + threadIdx.x;
  if (e >= NUM_EDGES) return;
  int src = ei[e], dst = ei[NUM_EDGES + e];
  const float* xl = xlr + (size_t)src * 16;
  const float* xr = xlr + (size_t)dst * 16 + 8;
  float s = 0.0f;
  #pragma unroll
  for (int k = 0; k < GAT_OUT; ++k) {
    float v = xl[k] + xr[k];
    v = (v > 0.0f) ? v : NEG_SLOPE * v;
    s += att[k] * v;
  }
  ev[e] = s;
  atomicMax(&mkey[dst], fkey(s));
}

// ---------- 7: exp(e - m[dst]) and segment sum over dst ----------
__global__ void k_numden(const int* __restrict__ ei, const float* __restrict__ ev,
                         const unsigned* __restrict__ mkey, float* num, float* den) {
  int e = blockIdx.x * blockDim.x + threadIdx.x;
  if (e >= NUM_EDGES) return;
  int dst = ei[NUM_EDGES + e];
  float nu = expf(ev[e] - keyf(mkey[dst]));
  num[e] = nu;
  atomicAdd(&den[dst], nu);
}

// ---------- 8: ce_space accumulation (one wave per edge, 32-class dot) ----------
__global__ void k_ce(const int* __restrict__ ei, const float* __restrict__ num,
                     const float* __restrict__ den, const float* __restrict__ P,
                     float* ceacc) {
  int gid = blockIdx.x * blockDim.x + threadIdx.x;
  int e = gid >> 5, lane = gid & 31;
  if (e >= NUM_EDGES) return;
  int src = ei[e], dst = ei[NUM_EDGES + e];
  float alpha = num[e] / (den[dst] + 1e-16f);
  float t = P[(size_t)src * NUM_CLASSES + lane] *
            logf(P[(size_t)dst * NUM_CLASSES + lane] + 1e-8f);
  t = waveSum(t);
  if (lane == 0) atomicAdd(ceacc, alpha * t);
}

// ---------- 9: finalize scalars ----------
__global__ void k_final(const float* accums, float* out) {
  if (threadIdx.x == 0 && blockIdx.x == 0) {
    out[0] = accums[0] / (float)N_SUB;
    out[1] = -accums[1] / (float)N_SUB;
  }
}

extern "C" void kernel_launch(void* const* d_in, const int* in_sizes, int n_in,
                              void* d_out, int out_size, void* d_ws, size_t ws_size,
                              hipStream_t stream) {
  (void)in_sizes; (void)n_in; (void)out_size; (void)ws_size;
  const float* x      = (const float*)d_in[0];
  const float* Mu     = (const float*)d_in[1];
  const float* Var    = (const float*)d_in[2];
  const int*   ei     = (const int*)d_in[3];
  const int*   subset = (const int*)d_in[4];
  const float* rate   = (const float*)d_in[5];
  const float* W      = (const float*)d_in[6];
  const float* S      = (const float*)d_in[7];
  const float* W_l    = (const float*)d_in[8];
  const float* b_l    = (const float*)d_in[9];
  const float* W_r    = (const float*)d_in[10];
  const float* b_r    = (const float*)d_in[11];
  const float* att    = (const float*)d_in[12];
  float* out = (float*)d_out;
  float* P   = out + 2;                    // [N_SUB, 32] output block

  float* ws = (float*)d_ws;
  const size_t OFF_ACC   = 0;                                   // N*G (also x_clean)
  const size_t OFF_XLR   = OFF_ACC  + (size_t)N_SUB * NUM_GENES;
  const size_t OFF_DEG   = OFF_XLR  + (size_t)N_SUB * 16;
  const size_t OFF_EV    = OFF_DEG  + N_SUB;
  const size_t OFF_NUM   = OFF_EV   + NUM_EDGES;
  const size_t OFF_MKEY  = OFF_NUM  + NUM_EDGES;
  const size_t OFF_DEN   = OFF_MKEY + N_SUB;
  const size_t OFF_DVEC  = OFF_DEN  + N_SUB;
  const size_t OFF_ACCUM = OFF_DVEC + 32;
  const size_t OFF_BVEC  = OFF_ACCUM + 8;
  size_t off_half = (OFF_BVEC + 16 + 15) & ~(size_t)15;         // 64B align for tables
  _Float16* gtab = (_Float16*)(ws + off_half);                  // invvB | muvB | lrB
  const int ZERO_N = (int)off_half;

  k_zero<<<(ZERO_N + 255) / 256, 256, 0, stream>>>(ws, ZERO_N);
  k_build_tabs<<<32, 256, 0, stream>>>(Mu, Var, W_l, W_r, b_l, b_r,
                                       gtab, ws + OFF_DVEC, ws + OFF_BVEC);
  k_spill_acc<<<(NUM_EDGES * 32 + 255) / 256, 256, 0, stream>>>(x, ei, ws + OFF_ACC, ws + OFF_DEG);
  k_xclean<<<(N_SUB * NUM_GENES + 255) / 256, 256, 0, stream>>>(x, ws + OFF_DEG, rate, ws + OFF_ACC);
  k_softmax<<<(N_SUB * 32 + 255) / 256, 256, 0, stream>>>(W, subset, P);
  k_wmma<<<(NTILES + 3) / 4, 128, 0, stream>>>(ws + OFF_ACC, gtab,
      ws + OFF_DVEC, ws + OFF_BVEC, subset, S, P, ws + OFF_XLR, ws + OFF_ACCUM);
  k_edge<<<(NUM_EDGES + 255) / 256, 256, 0, stream>>>(ei, ws + OFF_XLR, att,
      ws + OFF_EV, (unsigned*)(ws + OFF_MKEY));
  k_numden<<<(NUM_EDGES + 255) / 256, 256, 0, stream>>>(ei, ws + OFF_EV,
      (const unsigned*)(ws + OFF_MKEY), ws + OFF_NUM, ws + OFF_DEN);
  k_ce<<<(NUM_EDGES * 32 + 255) / 256, 256, 0, stream>>>(ei, ws + OFF_NUM,
      ws + OFF_DEN, P, ws + OFF_ACCUM + 1);
  k_final<<<1, 32, 0, stream>>>(ws + OFF_ACCUM, out);
}
